// WindowAttention_59442347377155
// MI455X (gfx1250) — matrix operand order repaired and checked by reading, not compiled, not run on hardware
//
#include <hip/hip_runtime.h>
#include <hip/hip_bf16.h>
#include <stdint.h>

typedef __attribute__((ext_vector_type(16))) __bf16 v16bf;
typedef __attribute__((ext_vector_type(8)))  float  v8f;

static constexpr int HEADS  = 8;
static constexpr int NTOK   = 49;     // tokens per window
static constexpr int NPAD   = 64;     // padded tokens
static constexpr int HD     = 32;     // head dim
static constexpr int DIM    = 256;
static constexpr int BATCH  = 2048;
static constexpr int MROWS  = BATCH * NTOK;     // 100352 (divisible by 16)
static constexpr int NWIN   = 64;
static constexpr float ATT_SCALE = 0.17677669529663687f;  // 32^-0.5

// Fragment: 16 bf16 values per lane = 8 VGPRs
union FragU {
  uint4 q[2];
  unsigned short s[16];
  v16bf v;
};

__device__ __forceinline__ unsigned short f2bf(float f) {
  union { float f; unsigned u; } c; c.f = f;
  unsigned u = c.u;
  u = u + 0x7FFFu + ((u >> 16) & 1u);   // round-to-nearest-even
  return (unsigned short)(u >> 16);
}

__device__ __forceinline__ unsigned pack2(float a, float b) {
  return (unsigned)f2bf(a) | ((unsigned)f2bf(b) << 16);
}

// ---- Kernel C: elementwise fp32 -> bf16 (8 elems / thread, b128 in/out) ----
__global__ __launch_bounds__(256) void k_cvt(const float* __restrict__ in,
                                             unsigned* __restrict__ out, int n8) {
  int t = blockIdx.x * blockDim.x + threadIdx.x;
  if (t >= n8) return;
  const float4* p = (const float4*)(in + (size_t)t * 8);
  float4 f0 = p[0], f1 = p[1];
  uint4 o;
  o.x = pack2(f0.x, f0.y);
  o.y = pack2(f0.z, f0.w);
  o.z = pack2(f1.x, f1.y);
  o.w = pack2(f1.z, f1.w);
  *(uint4*)(out + (size_t)t * 4) = o;
}

// ---- Kernel 0: gather relative position bias -> [H][49][49] fp32 ----
__global__ void k_biasgather(const float* __restrict__ bias_table,
                             const int* __restrict__ rel_index,
                             float* __restrict__ biasws) {
  int t = blockIdx.x * blockDim.x + threadIdx.x;
  if (t >= HEADS * NTOK * NTOK) return;
  int hh = t / (NTOK * NTOK);
  int r  = t - hh * (NTOK * NTOK);
  biasws[t] = bias_table[rel_index[r] * HEADS + hh];
}

// A-fragment (16x32 bf16) from bf16 row-major: two contiguous 16B loads
// lane<16: row=lr, K = k0+0..15; lane>=16: K = k0+16..31
__device__ __forceinline__ void loadA_bf(FragU& a, const unsigned short* arow,
                                         int k0, int kbase0) {
  const unsigned short* p = arow + k0 + 2 * kbase0;
  a.q[0] = *(const uint4*)p;
  a.q[1] = *(const uint4*)(p + 8);
}

// B-fragment (32x16 bf16) from bf16 weight row W[n][k]: 32B contiguous
// lane<16: col=lr, K=k0+0..15; lane>=16: K=k0+16..31
__device__ __forceinline__ void loadB_bf(FragU& b, const unsigned short* wrow) {
  b.q[0] = *(const uint4*)wrow;
  b.q[1] = *(const uint4*)(wrow + 8);
}

// ---- Kernel 1: qkv = x @ qkv_w^T + qkv_b, scatter to q/k/vT bf16 ----
__global__ __launch_bounds__(128) void k_qkv(
    const unsigned short* __restrict__ xbf, const unsigned short* __restrict__ wbf,
    const float* __restrict__ qkv_b,
    unsigned short* __restrict__ qws, unsigned short* __restrict__ kws,
    unsigned short* __restrict__ vws)
{
  const int lane = threadIdx.x & 31;
  const int wv   = threadIdx.x >> 5;
  const int lr   = lane & 15;
  const int half = lane >> 4;
  const int kbase0 = half ? 8 : 0;

  const int gw = blockIdx.x * 4 + wv;       // 75264 waves
  const int mt = gw / 12;                   // 6272 row tiles
  const int ng = gw % 12;                   // 12 groups of 64 of 768 cols
  const int m0 = mt * 16;
  const int n0 = ng * 64;

  v8f acc[4];
#pragma unroll
  for (int t = 0; t < 4; ++t)
#pragma unroll
    for (int i = 0; i < 8; ++i) acc[t][i] = 0.0f;

  const unsigned short* arow = xbf + (size_t)(m0 + lr) * DIM;
  const unsigned short* wbase = wbf + (size_t)(n0 + lr) * DIM + half * 16;
  for (int k0 = 0; k0 < DIM; k0 += 32) {
    if (k0 + 32 < DIM) __builtin_prefetch(arow + k0 + 32, 0, 1);
    FragU a;
    loadA_bf(a, arow, k0, kbase0);
#pragma unroll
    for (int t = 0; t < 4; ++t) {
      FragU b;
      loadB_bf(b, wbase + (size_t)t * 16 * DIM + k0);
      acc[t] = __builtin_amdgcn_wmma_f32_16x16x32_bf16(
          false, a.v, false, b.v, (short)0, acc[t], false, false);
    }
  }

#pragma unroll
  for (int t = 0; t < 4; ++t) {
    const int col = n0 + t * 16 + lr;
    const float bias = qkv_b[col];
    const int which = col >> 8;          // 0=q 1=k 2=v
    const int hh = (col >> 5) & 7;
    const int d  = col & 31;
#pragma unroll
    for (int i = 0; i < 8; ++i) {
      const int r = m0 + half * 8 + i;
      const int b = r / NTOK;
      const int n = r - b * NTOK;
      const float val = acc[t][i] + bias;
      const size_t bh = (size_t)b * HEADS + hh;
      if (which == 0)
        qws[(bh * NPAD + n) * HD + d] = f2bf(val * ATT_SCALE);
      else if (which == 1)
        kws[(bh * NPAD + n) * HD + d] = f2bf(val);
      else
        vws[(bh * HD + d) * NPAD + n] = f2bf(val);   // transposed for B-frags
    }
  }
}

// ---- Kernel 2: per-(b,h) attention: softmax(q k^T + bias + mask) v ----
__global__ __launch_bounds__(128) void k_attn(
    const unsigned short* __restrict__ qws, const unsigned short* __restrict__ kws,
    const unsigned short* __restrict__ vws, const float* __restrict__ biasws,
    const float* __restrict__ mask, unsigned short* __restrict__ obuf)
{
  __shared__ unsigned short plds[4][16][64];

  const int bh = blockIdx.x;
  const int b  = bh >> 3;
  const int h  = bh & 7;
  const int w  = b & (NWIN - 1);
  const unsigned short* qb = qws + (size_t)bh * NPAD * HD;
  const unsigned short* kb = kws + (size_t)bh * NPAD * HD;
  const unsigned short* vb = vws + (size_t)bh * HD * NPAD;

  const int lane = threadIdx.x & 31;
  const int wv   = threadIdx.x >> 5;         // row tile 0..3
  const int lr   = lane & 15;
  const int half = lane >> 4;
  const int kbase0 = half ? 8 : 0;

  // q A-fragment (16x32), single K step since hd==32
  FragU qa;
  {
    const unsigned short* p = qb + (wv * 16 + lr) * HD + 2 * kbase0;
    qa.q[0] = *(const uint4*)p;
    qa.q[1] = *(const uint4*)(p + 8);
  }

  v8f s[4];
#pragma unroll
  for (int t = 0; t < 4; ++t) {
#pragma unroll
    for (int i = 0; i < 8; ++i) s[t][i] = 0.0f;
    FragU kf;
    const unsigned short* p = kb + (t * 16 + lr) * HD + half * 16;
    kf.q[0] = *(const uint4*)p;
    kf.q[1] = *(const uint4*)(p + 8);
    s[t] = __builtin_amdgcn_wmma_f32_16x16x32_bf16(
        false, qa.v, false, kf.v, (short)0, s[t], false, false);
  }

  // bias + shift mask + padding mask (q is pre-scaled)
  float sv[4][8];
#pragma unroll
  for (int t = 0; t < 4; ++t) {
    const int nk = t * 16 + lr;
#pragma unroll
    for (int i = 0; i < 8; ++i) {
      const int nq = wv * 16 + half * 8 + i;
      float val = s[t][i];
      if (nk < NTOK && nq < NTOK)
        val += biasws[(h * NTOK + nq) * NTOK + nk] +
               mask[((size_t)w * NTOK + nq) * NTOK + nk];
      else
        val = -1e30f;
      sv[t][i] = val;
    }
  }

  // row softmax: reduce across 4 tiles + 16 lanes of each half
#pragma unroll
  for (int i = 0; i < 8; ++i) {
    float m = sv[0][i];
#pragma unroll
    for (int t = 1; t < 4; ++t) m = fmaxf(m, sv[t][i]);
#pragma unroll
    for (int o = 8; o >= 1; o >>= 1) m = fmaxf(m, __shfl_xor(m, o, 32));
    float su = 0.0f;
#pragma unroll
    for (int t = 0; t < 4; ++t) { float e = __expf(sv[t][i] - m); sv[t][i] = e; su += e; }
#pragma unroll
    for (int o = 8; o >= 1; o >>= 1) su += __shfl_xor(su, o, 32);
    const float inv = 1.0f / su;
#pragma unroll
    for (int t = 0; t < 4; ++t) sv[t][i] *= inv;
  }

  // stage P through LDS to re-stripe into A-fragments
#pragma unroll
  for (int t = 0; t < 4; ++t)
#pragma unroll
    for (int i = 0; i < 8; ++i)
      plds[wv][half * 8 + i][t * 16 + lr] = f2bf(sv[t][i]);
  __syncthreads();

  // O = P(16x64) @ v(64x32): K steps {0,32}, two 16-wide d tiles
  v8f o[2];
#pragma unroll
  for (int dt = 0; dt < 2; ++dt)
#pragma unroll
    for (int i = 0; i < 8; ++i) o[dt][i] = 0.0f;

#pragma unroll
  for (int kk = 0; kk < 64; kk += 32) {
    FragU pa;
    const unsigned short* pp = &plds[wv][lr][kk + 2 * kbase0];
    pa.q[0] = *(const uint4*)pp;
    pa.q[1] = *(const uint4*)(pp + 8);
#pragma unroll
    for (int dt = 0; dt < 2; ++dt) {
      FragU vf;
      const unsigned short* vp = vb + (dt * 16 + lr) * NPAD + kk + half * 16;
      vf.q[0] = *(const uint4*)vp;
      vf.q[1] = *(const uint4*)(vp + 8);
      o[dt] = __builtin_amdgcn_wmma_f32_16x16x32_bf16(
          false, pa.v, false, vf.v, (short)0, o[dt], false, false);
    }
  }

#pragma unroll
  for (int dt = 0; dt < 2; ++dt)
#pragma unroll
    for (int i = 0; i < 8; ++i) {
      const int nq = wv * 16 + half * 8 + i;
      if (nq < NTOK)
        obuf[((size_t)(b * NTOK + nq)) * DIM + h * HD + dt * 16 + lr] = f2bf(o[dt][i]);
    }
}

// ---- Kernel 3: out = O @ proj_w^T + proj_b (all-bf16 operands) ----
__global__ __launch_bounds__(128) void k_proj(
    const unsigned short* __restrict__ obuf, const unsigned short* __restrict__ wbf,
    const float* __restrict__ proj_b, float* __restrict__ out)
{
  const int lane = threadIdx.x & 31;
  const int wv   = threadIdx.x >> 5;
  const int lr   = lane & 15;
  const int half = lane >> 4;
  const int kbase0 = half ? 8 : 0;

  const int gw = blockIdx.x * 4 + wv;       // 25088 waves
  const int mt = gw >> 2;
  const int ng = gw & 3;
  const int m0 = mt * 16;
  const int n0 = ng * 64;

  v8f acc[4];
#pragma unroll
  for (int t = 0; t < 4; ++t)
#pragma unroll
    for (int i = 0; i < 8; ++i) acc[t][i] = 0.0f;

  const unsigned short* arow = obuf + (size_t)(m0 + lr) * DIM;
  const unsigned short* wbase = wbf + (size_t)(n0 + lr) * DIM + half * 16;
  for (int k0 = 0; k0 < DIM; k0 += 32) {
    if (k0 + 32 < DIM) __builtin_prefetch(arow + k0 + 32, 0, 1);
    FragU a;
    loadA_bf(a, arow, k0, kbase0);
#pragma unroll
    for (int t = 0; t < 4; ++t) {
      FragU b;
      loadB_bf(b, wbase + (size_t)t * 16 * DIM + k0);
      acc[t] = __builtin_amdgcn_wmma_f32_16x16x32_bf16(
          false, a.v, false, b.v, (short)0, acc[t], false, false);
    }
  }

#pragma unroll
  for (int t = 0; t < 4; ++t) {
    const int col = n0 + t * 16 + lr;
    const float bias = proj_b[col];
#pragma unroll
    for (int i = 0; i < 8; ++i) {
      const int r = m0 + half * 8 + i;
      out[(size_t)r * DIM + col] = acc[t][i] + bias;
    }
  }
}

extern "C" void kernel_launch(void* const* d_in, const int* in_sizes, int n_in,
                              void* d_out, int out_size, void* d_ws, size_t ws_size,
                              hipStream_t stream) {
  const float* x          = (const float*)d_in[0];
  const float* mask       = (const float*)d_in[1];
  const float* qkv_w      = (const float*)d_in[2];
  const float* qkv_b      = (const float*)d_in[3];
  const float* bias_table = (const float*)d_in[4];
  const float* proj_w     = (const float*)d_in[5];
  const float* proj_b     = (const float*)d_in[6];
  const int*   rel_index  = (const int*)d_in[7];

  char* base = (char*)d_ws;
  size_t off = 0;
  float* biasws = (float*)(base + off);            off += 77056;                  // 8*49*49 f32 (padded)
  unsigned short* xbf   = (unsigned short*)(base + off); off += (size_t)MROWS * DIM * 2;   // x bf16
  unsigned short* wqkvb = (unsigned short*)(base + off); off += (size_t)3 * DIM * DIM * 2; // qkv_w bf16
  unsigned short* wprjb = (unsigned short*)(base + off); off += (size_t)DIM * DIM * 2;     // proj_w bf16
  unsigned short* qws   = (unsigned short*)(base + off); off += (size_t)BATCH * HEADS * NPAD * HD * 2;
  unsigned short* kws   = (unsigned short*)(base + off); off += (size_t)BATCH * HEADS * NPAD * HD * 2;
  unsigned short* vws   = (unsigned short*)(base + off); off += (size_t)BATCH * HEADS * HD * NPAD * 2;
  unsigned short* obuf  = (unsigned short*)(base + off); off += (size_t)MROWS * DIM * 2;

  // one-time fp32->bf16 conversions (hot loops then run pure bf16)
  const int x8 = MROWS * DIM / 8, wq8 = 3 * DIM * DIM / 8, wp8 = DIM * DIM / 8;
  k_cvt<<<(x8 + 255) / 256, 256, 0, stream>>>(x, (unsigned*)xbf, x8);
  k_cvt<<<(wq8 + 255) / 256, 256, 0, stream>>>(qkv_w, (unsigned*)wqkvb, wq8);
  k_cvt<<<(wp8 + 255) / 256, 256, 0, stream>>>(proj_w, (unsigned*)wprjb, wp8);
  k_biasgather<<<(HEADS * NTOK * NTOK + 255) / 256, 256, 0, stream>>>(
      bias_table, rel_index, biasws);

  k_qkv<<<(MROWS / 16) * 12 / 4, 128, 0, stream>>>(xbf, wqkvb, qkv_b, qws, kws, vws);
  k_attn<<<BATCH * HEADS, 128, 0, stream>>>(qws, kws, vws, biasws, mask, obuf);
  k_proj<<<(MROWS / 16) * 4 / 4, 128, 0, stream>>>(obuf, wprjb, proj_b, (float*)d_out);
}
